// DecoderRNNT_62534723829919
// MI455X (gfx1250) — compile-verified, matrix operand order checked by reading, not run on hardware
//
#include <hip/hip_runtime.h>
#include <hip/hip_bf16.h>

// ---------------------------------------------------------------------------
// RNN-T decoder + joint for MI455X (gfx1250, wave32, WMMA, TDM).
//   - all GEMMs via v_wmma_f32_16x16x32_bf16 (f32 accumulate)
//   - all weights pre-packed to fragment-major bf16 (coalesced 1KB/wave loads)
//   - joint kernel double-buffers W_out K-chunks in LDS via tensor_load_to_lds
//     (TDM): DMA of chunk c+1 overlaps tanh-build + WMMA of chunk c, with a
//     counted s_wait_tensorcnt(1); the 100M-element z tensor never touches HBM
//   - persistent 16-WG LSTM with LDS-resident cell state + grid barrier
// ---------------------------------------------------------------------------

typedef __bf16 bf16_t;
typedef __attribute__((ext_vector_type(16))) __bf16 v16bf;
typedef __attribute__((ext_vector_type(8)))  float  v8f;
typedef __attribute__((ext_vector_type(4)))  unsigned int u32x4;
typedef __attribute__((ext_vector_type(4)))  int i32x4;
typedef __attribute__((ext_vector_type(8)))  int i32x8;

#define B_    16
#define T_    200
#define U_    64
#define E_    512
#define D_    512
#define J_    512
#define ODIM_ 600
#define OPAD_ 608      // 38 tiles of 16
#define NT_O  38       // OPAD_/16
#define G_    2048     // 4*D gates
// per-K-chunk packed W_out block: 38 tiles * 32 lanes * 16 bf16
#define CHUNK_ELEMS (NT_O * 32 * 16)      // 19456 bf16
#define CHUNK_BYTES (CHUNK_ELEMS * 2)     // 38912 B
#define CHUNK_QW    (CHUNK_BYTES / 8)     // 4864 x 8B
#define NCHUNK      (J_ >> 5)             // 16

__device__ __forceinline__ float sigmoidf_(float x) { return 1.0f / (1.0f + __expf(-x)); }

__device__ __forceinline__ v8f wmma_bf16(v16bf a, v16bf b, v8f c) {
  return __builtin_amdgcn_wmma_f32_16x16x32_bf16(
      false, a, false, b, (short)0, c, false, false);
}

// Row-major fragment load (A operands): lane l -> row row_base + l%16,
// K offset k_base + (l/16)*16.  32B per lane.
__device__ __forceinline__ v16bf load_frag(const bf16_t* p, int ld, int row_base, int k_base) {
  const int lane = threadIdx.x & 31;
  const bf16_t* src = p + (long)(row_base + (lane & 15)) * ld + k_base + ((lane >> 4) << 4);
  return *reinterpret_cast<const v16bf*>(src);
}

// Fragment-major load (packed B operands): one contiguous 1KB block per frag.
__device__ __forceinline__ v16bf load_frag_packed(const bf16_t* p, long frag_idx) {
  const int lane = threadIdx.x & 31;
  return *reinterpret_cast<const v16bf*>(p + (frag_idx * 32 + lane) * 16);
}

// Issue a 1-D TDM copy of one packed W_out chunk (38,912 B) into LDS.
__device__ __forceinline__ void tdm_issue(const bf16_t* gsrc, unsigned lds_off) {
  const unsigned long long ga = (unsigned long long)(size_t)gsrc;
  u32x4 g0;
  g0.x = 1u;                                                 // count=1 user D#
  g0.y = lds_off;                                            // LDS dest (bytes)
  g0.z = (unsigned)(ga & 0xffffffffu);                       // global_addr lo
  g0.w = (unsigned)((ga >> 32) & 0x01ffffffu) | (2u << 30);  // addr hi | type=2
  i32x8 g1 = (i32x8)0;
  g1[0] = (3 << 16);                 // data_size = 8B
  g1[1] = (int)(CHUNK_QW << 16);     // tensor_dim0 lo16 @ dw1[31:16]
  g1[2] = (1 << 16);                 // tensor_dim1 = 1
  g1[3] = (int)(CHUNK_QW << 16);     // tile_dim0 @ dw3[31:16]
  g1[5] = (int)CHUNK_QW;             // tensor_dim0_stride lo32
  i32x4 z4 = (i32x4)0;
#if defined(__clang_major__) && (__clang_major__ >= 23)
  i32x8 z8 = (i32x8)0;
  __builtin_amdgcn_tensor_load_to_lds(g0, g1, z4, z4, z8, 0);
#else
  __builtin_amdgcn_tensor_load_to_lds(g0, g1, z4, z4, 0);
#endif
}

// ------------------------------ prep kernels -------------------------------

__global__ void k_cvt_bf16(const float* __restrict__ src, bf16_t* __restrict__ dst, long n) {
  long i = (long)blockIdx.x * blockDim.x + threadIdx.x;
  long stride = (long)gridDim.x * blockDim.x;
  for (; i < n; i += stride) dst[i] = (bf16_t)src[i];
}

// Pack f32 [N][K] weight matrix into bf16 fragment-major layout.
//  tile-major  (chunk_major=0): frag_idx = n0*(K/32) + c
//  chunk-major (chunk_major=1): frag_idx = c*(N/16) + n0   (contiguous per K chunk)
// Rows >= Nvalid are zero-padded.
__global__ void k_pack_frag(const float* __restrict__ src, bf16_t* __restrict__ dst,
                            int N, int K, int Nvalid, int chunk_major) {
  const int NT = N >> 4, NC = K >> 5;
  const long total = (long)N * K;
  long i = (long)blockIdx.x * blockDim.x + threadIdx.x;
  const long stride = (long)gridDim.x * blockDim.x;
  for (; i < total; i += stride) {
    const int e    = (int)(i & 15);
    const int lane = (int)((i >> 4) & 31);
    const long rest = i >> 9;
    int n0, c;
    if (chunk_major) { c = (int)(rest / NT); n0 = (int)(rest % NT); }
    else             { n0 = (int)(rest / NC); c = (int)(rest % NC); }
    const int row = n0 * 16 + (lane & 15);
    const int k   = c * 32 + ((lane >> 4) << 4) + e;
    const float v = (row < Nvalid) ? src[(long)row * K + k] : 0.0f;
    dst[i] = (bf16_t)v;
  }
}

__global__ void k_bias_add(const float* __restrict__ a, const float* __restrict__ b,
                           float* __restrict__ dst, int n) {
  int i = blockIdx.x * blockDim.x + threadIdx.x;
  if (i < n) dst[i] = a[i] + b[i];
}

__global__ void k_embed(const int* __restrict__ ys, const float* __restrict__ table,
                        bf16_t* __restrict__ eys) {
  int bu = blockIdx.x;
  int tok = ys[bu];
  const float* src = table + (long)tok * E_;
  bf16_t* dst = eys + (long)bu * E_;
  for (int e = threadIdx.x; e < E_; e += blockDim.x) dst[e] = (bf16_t)src[e];
}

__global__ void k_init_state(bf16_t* __restrict__ h0, bf16_t* __restrict__ h1,
                             unsigned* __restrict__ bar) {
  int i = blockIdx.x * blockDim.x + threadIdx.x;
  const int n = 2 * B_ * D_;
  if (i < n) { h0[i] = (bf16_t)0.0f; h1[i] = (bf16_t)0.0f; }
  if (i == 0) *bar = 0u;
}

// ------------------------- generic bf16 WMMA GEMM --------------------------
// C[M x N](f32) = A[M x K](bf16 row-major) * Bp(packed tile-major)^T + bias
// grid = (M/32, N/64), block = 256 (8 waves -> 2x4 tiles of 16x16)

__global__ void __launch_bounds__(256)
k_gemm_bf16(const bf16_t* __restrict__ A, const bf16_t* __restrict__ Bp,
            const float* __restrict__ bias, float* __restrict__ C,
            int M, int N, int K) {
  const int wave = threadIdx.x >> 5;
  const int lane = threadIdx.x & 31;
  const int m_base = blockIdx.x * 32 + (wave >> 2) * 16;
  const int n_base = blockIdx.y * 64 + (wave & 3) * 16;
  const int NC = K >> 5;
  const long fb0 = (long)(n_base >> 4) * NC;

  v8f acc = {};
  for (int kc = 0; kc < K; kc += 32) {
    v16bf a = load_frag(A, K, m_base, kc);
    v16bf b = load_frag_packed(Bp, fb0 + (kc >> 5));
    // pull next packed B fragment into the WGP cache (locality 3 -> WGP scope)
    __builtin_prefetch(Bp + ((fb0 + (kc >> 5) + 1) * 32 + lane) * 16, 0, 3);
    acc = wmma_bf16(a, b, acc);
  }
  const int n    = n_base + (lane & 15);
  const int mrow = m_base + ((lane >> 4) << 3);
  const float bv = bias ? bias[n] : 0.0f;
#pragma unroll
  for (int v = 0; v < 8; ++v)
    C[(long)(mrow + v) * N + n] = acc[v] + bv;
}

// ------------------------------- LSTM ---------------------------------------

__device__ __forceinline__ void grid_barrier(unsigned* bar, unsigned nwg,
                                             unsigned& phase, int tid) {
  __threadfence();
  __syncthreads();
  __builtin_amdgcn_s_cluster_barrier();    // NOP when not launched as a cluster
  ++phase;
  if (tid == 0) {
    __hip_atomic_fetch_add(bar, 1u, __ATOMIC_ACQ_REL, __HIP_MEMORY_SCOPE_AGENT);
    while (__hip_atomic_load(bar, __ATOMIC_ACQUIRE, __HIP_MEMORY_SCOPE_AGENT) < nwg * phase)
      __builtin_amdgcn_s_sleep(1);
  }
  __syncthreads();
  __threadfence();
}

// 16 persistent WGs; WG w owns hidden slice [w*32, w*32+32) of BOTH layers.
__global__ void __launch_bounds__(256)
k_lstm(const bf16_t* __restrict__ eys,                        // [B][U][E]
       const bf16_t* __restrict__ Wih0p, const bf16_t* __restrict__ Whh0p,  // packed
       const bf16_t* __restrict__ Wih1p, const bf16_t* __restrict__ Whh1p,  // packed
       const float*  __restrict__ b0,    const float*  __restrict__ b1,     // [G_]
       bf16_t* __restrict__ h0buf, bf16_t* __restrict__ h1buf,              // [2][B][D]
       bf16_t* __restrict__ hdec,                                           // [B][U][D]
       unsigned* __restrict__ bar) {
  const int wg   = blockIdx.x;
  const int tid  = threadIdx.x;
  const int wave = tid >> 5;
  const int lane = tid & 31;
  const int type = wave >> 1;             // 0=i 1=f 2=g 3=o
  const int half = wave & 1;
  const int gate_row = type * D_ + wg * 32 + half * 16;
  const long gfrag   = (long)(gate_row >> 4) * (D_ >> 5);  // tile-major frag base

  __shared__ float gbuf[4][16][32];
  __shared__ float c0s[16][32];
  __shared__ float c1s[16][32];
  for (int i = tid; i < 16 * 32; i += 256) {
    (&c0s[0][0])[i] = 0.0f;
    (&c1s[0][0])[i] = 0.0f;
  }
  unsigned phase = 0;
  __syncthreads();

  for (int u = 0; u < U_; ++u) {
    const bf16_t* h0_prev = h0buf + (size_t)((u ^ 1) & 1) * (B_ * D_);
    bf16_t*       h0_cur  = h0buf + (size_t)(u & 1)       * (B_ * D_);
    const bf16_t* h1_prev = h1buf + (size_t)((u ^ 1) & 1) * (B_ * D_);
    bf16_t*       h1_cur  = h1buf + (size_t)(u & 1)       * (B_ * D_);

    // layer 0: gates = [ey_u | h0_prev] x [Wih0|Whh0]^T
    {
      v8f acc = {};
      const int m  = lane & 15;
      const int kh = (lane >> 4) << 4;
      for (int kc = 0; kc < E_; kc += 32) {
        const bf16_t* ap = eys + ((long)m * U_ + u) * E_ + kc + kh;
        v16bf a = *reinterpret_cast<const v16bf*>(ap);
        v16bf b = load_frag_packed(Wih0p, gfrag + (kc >> 5));
        __builtin_prefetch(Wih0p + ((gfrag + (kc >> 5) + 1) * 32 + lane) * 16, 0, 3);
        acc = wmma_bf16(a, b, acc);
      }
      for (int kc = 0; kc < D_; kc += 32) {
        v16bf a = load_frag(h0_prev, D_, 0, kc);
        v16bf b = load_frag_packed(Whh0p, gfrag + (kc >> 5));
        __builtin_prefetch(Whh0p + ((gfrag + (kc >> 5) + 1) * 32 + lane) * 16, 0, 3);
        acc = wmma_bf16(a, b, acc);
      }
      const int n    = lane & 15;
      const int mtop = (lane >> 4) << 3;
      const float bv = b0[gate_row + n];
#pragma unroll
      for (int v = 0; v < 8; ++v)
        gbuf[type][mtop + v][half * 16 + n] = acc[v] + bv;
    }
    __syncthreads();
    {
      int idx = tid * 2;
#pragma unroll
      for (int s = 0; s < 2; ++s, ++idx) {
        const int bi = idx >> 5, d = idx & 31;
        const float ig = sigmoidf_(gbuf[0][bi][d]);
        const float fg = sigmoidf_(gbuf[1][bi][d]);
        const float gg = tanhf(gbuf[2][bi][d]);
        const float og = sigmoidf_(gbuf[3][bi][d]);
        const float c  = fg * c0s[bi][d] + ig * gg;
        c0s[bi][d] = c;
        h0_cur[bi * D_ + wg * 32 + d] = (bf16_t)(og * tanhf(c));
      }
    }
    grid_barrier(bar, 16, phase, tid);

    // layer 1: gates = [h0_cur | h1_prev] x [Wih1|Whh1]^T
    {
      v8f acc = {};
      for (int kc = 0; kc < D_; kc += 32) {
        v16bf a = load_frag(h0_cur, D_, 0, kc);
        v16bf b = load_frag_packed(Wih1p, gfrag + (kc >> 5));
        __builtin_prefetch(Wih1p + ((gfrag + (kc >> 5) + 1) * 32 + lane) * 16, 0, 3);
        acc = wmma_bf16(a, b, acc);
      }
      for (int kc = 0; kc < D_; kc += 32) {
        v16bf a = load_frag(h1_prev, D_, 0, kc);
        v16bf b = load_frag_packed(Whh1p, gfrag + (kc >> 5));
        __builtin_prefetch(Whh1p + ((gfrag + (kc >> 5) + 1) * 32 + lane) * 16, 0, 3);
        acc = wmma_bf16(a, b, acc);
      }
      const int n    = lane & 15;
      const int mtop = (lane >> 4) << 3;
      const float bv = b1[gate_row + n];
#pragma unroll
      for (int v = 0; v < 8; ++v)
        gbuf[type][mtop + v][half * 16 + n] = acc[v] + bv;
    }
    __syncthreads();
    {
      int idx = tid * 2;
#pragma unroll
      for (int s = 0; s < 2; ++s, ++idx) {
        const int bi = idx >> 5, d = idx & 31;
        const float ig = sigmoidf_(gbuf[0][bi][d]);
        const float fg = sigmoidf_(gbuf[1][bi][d]);
        const float gg = tanhf(gbuf[2][bi][d]);
        const float og = sigmoidf_(gbuf[3][bi][d]);
        const float c  = fg * c1s[bi][d] + ig * gg;
        c1s[bi][d] = c;
        const float h = og * tanhf(c);
        h1_cur[bi * D_ + wg * 32 + d] = (bf16_t)h;
        hdec[((long)bi * U_ + u) * D_ + wg * 32 + d] = (bf16_t)h;
      }
    }
    grid_barrier(bar, 16, phase, tid);
  }
}

// ------------------------------ fused joint ---------------------------------
// out[b,t,u,o] = tanh(enc_p[b,t,:] + dec_p[b,u,:]) @ Wout^T + b_out
// One WG per (b, t, u-tile of 16).  Double-buffered TDM pipeline: while all
// waves run tanh-build + 38 WMMAs on chunk c (LDS buf c&1), wave 0's TDM
// streams chunk c+1 into the other buffer; counted s_wait_tensorcnt(1)
// guarantees chunk c has landed (TDM ops of a wave complete in order).
// Dynamic LDS: 2 * 38,912 B W buffers + 1 KB z buffer = 78,848 B.
__global__ void __launch_bounds__(256)
k_joint(const float* __restrict__ enc_p,   // [B][T][J]
        const float* __restrict__ dec_p,   // [B][U][J]
        const bf16_t* __restrict__ WoutPk, // chunk-major packed [16 chunks][38 frags]
        const float* __restrict__ b_out,   // [ODIM_]
        float* __restrict__ out) {         // [B][T][U][ODIM_]
  const int t  = blockIdx.x;
  const int ub = blockIdx.y * 16;
  const int b  = blockIdx.z;
  const int tid = threadIdx.x, wave = tid >> 5, lane = tid & 31;

  extern __shared__ __align__(64) char kj_smem[];
  bf16_t* wb0 = (bf16_t*)kj_smem;
  bf16_t* wb1 = (bf16_t*)(kj_smem + CHUNK_BYTES);
  bf16_t* zb  = (bf16_t*)(kj_smem + 2 * CHUNK_BYTES);   // [16][32]

  const float* enc_row  = enc_p + ((long)b * T_ + t) * J_;
  const float* dec_rows = dec_p + ((long)b * U_ + ub) * J_;
  const unsigned lds0 = (unsigned)(size_t)(void*)wb0;
  const unsigned lds1 = (unsigned)(size_t)(void*)wb1;

  v8f acc[5];
  v8f zero = {};
#pragma unroll
  for (int i = 0; i < 5; ++i) acc[i] = zero;

  // prologue: start DMA of chunk 0
  if (wave == 0) tdm_issue(WoutPk, lds0);

  for (int c = 0; c < NCHUNK; ++c) {
    // issue DMA for chunk c+1 into the other buffer (its readers from
    // iteration c-1 finished at the tail barrier)
    if (wave == 0 && (c + 1) < NCHUNK)
      tdm_issue(WoutPk + (size_t)(c + 1) * CHUNK_ELEMS, ((c + 1) & 1) ? lds1 : lds0);

    // build z chunk (overlaps both outstanding TDM transfers)
    {
      const int kc = c << 5;
      const int idx = tid * 2;
      const int ur = idx >> 5, j = idx & 31;
      const float* dr = dec_rows + (long)ur * J_ + kc;
      zb[ur * 32 + j]     = (bf16_t)tanhf(enc_row[kc + j]     + dr[j]);
      zb[ur * 32 + j + 1] = (bf16_t)tanhf(enc_row[kc + j + 1] + dr[j + 1]);
    }
    // chunk c arrived when at most the c+1 transfer is still in flight
    if (wave == 0) {
      if ((c + 1) < NCHUNK) __builtin_amdgcn_s_wait_tensorcnt(1);
      else                  __builtin_amdgcn_s_wait_tensorcnt(0);
    }
    __syncthreads();                       // wbuf[c&1] + zbuf ready

    const bf16_t* wbuf = (c & 1) ? wb1 : wb0;
    v16bf a;
    {
      const int m = lane & 15, kh = (lane >> 4) << 4;
      a = *reinterpret_cast<const v16bf*>(&zb[m * 32 + kh]);
    }
#pragma unroll
    for (int i = 0; i < 5; ++i) {
      const int nt = wave * 5 + i;         // wave-uniform guard: EXEC stays full
      if (nt < NT_O) {
        v16bf bfr = *reinterpret_cast<const v16bf*>(&wbuf[(nt * 32 + lane) * 16]);
        acc[i] = wmma_bf16(a, bfr, acc[i]);
      }
    }
    __syncthreads();                       // protect wbuf/zbuf before overwrite
  }

#pragma unroll
  for (int i = 0; i < 5; ++i) {
    const int nt = wave * 5 + i;
    if (nt >= NT_O) continue;
    const int n = nt * 16 + (lane & 15);
    if (n >= ODIM_) continue;              // after all WMMAs; divergence OK
    const float bv = b_out[n];
    const int mtop = (lane >> 4) << 3;
#pragma unroll
    for (int v = 0; v < 8; ++v) {
      const int u = ub + mtop + v;
      out[(((long)b * T_ + t) * U_ + u) * ODIM_ + n] = acc[i][v] + bv;
    }
  }
}

// ------------------------------ launch --------------------------------------

extern "C" void kernel_launch(void* const* d_in, const int* in_sizes, int n_in,
                              void* d_out, int out_size, void* d_ws, size_t ws_size,
                              hipStream_t stream) {
  (void)in_sizes; (void)n_in; (void)out_size; (void)ws_size;
  const float* hs    = (const float*)d_in[0];
  const int*   ys    = (const int*)  d_in[1];
  const float* embed = (const float*)d_in[2];
  const float* Wih0  = (const float*)d_in[3];
  const float* Whh0  = (const float*)d_in[4];
  const float* bih0  = (const float*)d_in[5];
  const float* bhh0  = (const float*)d_in[6];
  const float* Wih1  = (const float*)d_in[7];
  const float* Whh1  = (const float*)d_in[8];
  const float* bih1  = (const float*)d_in[9];
  const float* bhh1  = (const float*)d_in[10];
  const float* Wenc  = (const float*)d_in[11];
  const float* benc  = (const float*)d_in[12];
  const float* Wdec  = (const float*)d_in[13];
  const float* Wout  = (const float*)d_in[14];
  const float* bout  = (const float*)d_in[15];

  char* ws = (char*)d_ws;
  size_t off = 0;
  auto alloc = [&](size_t bytes) -> char* {
    char* p = ws + off;
    off = (off + bytes + 255) & ~(size_t)255;
    return p;
  };
  bf16_t* wih0_pk = (bf16_t*)alloc((size_t)G_ * D_ * 2);
  bf16_t* whh0_pk = (bf16_t*)alloc((size_t)G_ * D_ * 2);
  bf16_t* wih1_pk = (bf16_t*)alloc((size_t)G_ * D_ * 2);
  bf16_t* whh1_pk = (bf16_t*)alloc((size_t)G_ * D_ * 2);
  bf16_t* wenc_pk = (bf16_t*)alloc((size_t)J_ * E_ * 2);
  bf16_t* wdec_pk = (bf16_t*)alloc((size_t)J_ * D_ * 2);
  bf16_t* wout_pk = (bf16_t*)alloc((size_t)OPAD_ * J_ * 2);
  bf16_t* hs_bf   = (bf16_t*)alloc((size_t)B_ * T_ * E_ * 2);
  bf16_t* eys_bf  = (bf16_t*)alloc((size_t)B_ * U_ * E_ * 2);
  float*  b0      = (float*) alloc((size_t)G_ * 4);
  float*  b1      = (float*) alloc((size_t)G_ * 4);
  bf16_t* h0buf   = (bf16_t*)alloc((size_t)2 * B_ * D_ * 2);
  bf16_t* h1buf   = (bf16_t*)alloc((size_t)2 * B_ * D_ * 2);
  bf16_t* hdec_bf = (bf16_t*)alloc((size_t)B_ * U_ * D_ * 2);
  float*  enc_p   = (float*) alloc((size_t)B_ * T_ * J_ * 4);
  float*  dec_p   = (float*) alloc((size_t)B_ * U_ * J_ * 4);
  unsigned* bar   = (unsigned*)alloc(256);

  // ---- prep: fragment packing (f32 -> bf16), embed gather, state init
  k_pack_frag<<<256, 256, 0, stream>>>(Wih0, wih0_pk, G_, E_, G_, 0);
  k_pack_frag<<<256, 256, 0, stream>>>(Whh0, whh0_pk, G_, D_, G_, 0);
  k_pack_frag<<<256, 256, 0, stream>>>(Wih1, wih1_pk, G_, D_, G_, 0);
  k_pack_frag<<<256, 256, 0, stream>>>(Whh1, whh1_pk, G_, D_, G_, 0);
  k_pack_frag<<<256, 256, 0, stream>>>(Wenc, wenc_pk, J_, E_, J_, 0);
  k_pack_frag<<<256, 256, 0, stream>>>(Wdec, wdec_pk, J_, D_, J_, 0);
  k_pack_frag<<<256, 256, 0, stream>>>(Wout, wout_pk, OPAD_, J_, ODIM_, 1);
  k_cvt_bf16<<<256, 256, 0, stream>>>(hs, hs_bf, (long)B_ * T_ * E_);
  k_bias_add<<<(G_ + 255) / 256, 256, 0, stream>>>(bih0, bhh0, b0, G_);
  k_bias_add<<<(G_ + 255) / 256, 256, 0, stream>>>(bih1, bhh1, b1, G_);
  k_embed<<<B_ * U_, 256, 0, stream>>>(ys, embed, eys_bf);
  k_init_state<<<(2 * B_ * D_ + 255) / 256, 256, 0, stream>>>(h0buf, h1buf, bar);

  // ---- enc projection: (B*T=3200) x 512, K=512
  k_gemm_bf16<<<dim3(B_ * T_ / 32, J_ / 64), 256, 0, stream>>>(
      hs_bf, wenc_pk, benc, enc_p, B_ * T_, J_, E_);

  // ---- persistent 2-layer LSTM over U=64 steps
  k_lstm<<<16, 256, 0, stream>>>(eys_bf, wih0_pk, whh0_pk, wih1_pk, whh1_pk,
                                 b0, b1, h0buf, h1buf, hdec_bf, bar);

  // ---- dec projection: (B*U=1024) x 512, K=512 (no bias)
  k_gemm_bf16<<<dim3(B_ * U_ / 32, J_ / 64), 256, 0, stream>>>(
      hdec_bf, wdec_pk, nullptr, dec_p, B_ * U_, J_, D_);

  // ---- fused tanh + output GEMM with double-buffered TDM W_out staging
  const size_t joint_lds = 2 * CHUNK_BYTES + 16 * 32 * sizeof(bf16_t);
  k_joint<<<dim3(T_, U_ / 16, B_), 256, joint_lds, stream>>>(
      enc_p, dec_p, wout_pk, bout, (float*)d_out);
}
